// MixtureOfExperts_81484119540363
// MI455X (gfx1250) — matrix operand order repaired and checked by reading, not compiled, not run on hardware
//
#include <hip/hip_runtime.h>
#include <hip/hip_bf16.h>
#include <math.h>

// ---------------- problem constants (from reference setup) ----------------
#define NTOK 4096   // B*S = 2*2048 tokens
#define DM   2048   // model dim D
#define ID   1024   // expert hidden dim I
#define NE   8      // experts
#define TOPK 2
#define NTILES 8    // 16x16 output tiles per wave (wave tile = 16 x 128)

typedef __attribute__((ext_vector_type(16))) _Float16 v16h;
typedef __attribute__((ext_vector_type(8)))  _Float16 v8h;
typedef __attribute__((ext_vector_type(8)))  float    v8f;

union FragU { v16h v; v8h h[2]; };

// A-matrix 16x32 f16 fragment (ISA 7.12.2 layout):
// lanes 0-15: row M=lane, K = k..k+7 (VGPR0-3) and k+16..k+23 (VGPR4-7)
// lanes 16-31: row M=lane-16, K = k+8..k+15 and k+24..k+31
__device__ __forceinline__ v16h load_a_frag(const _Float16* __restrict__ rowptr,
                                            int k, int lane) {
    int koff = k + ((lane >> 4) << 3);
    FragU f;
    f.h[0] = *(const v8h*)(rowptr + koff);
    f.h[1] = *(const v8h*)(rowptr + koff + 16);
    return f.v;
}

// B-matrix 32x16 f16 fragment (same K order as A):
// lane n (0-15): col N=n, K = k..k+7 and k+16..k+23; lane n+16: K = k+8..k+15 and k+24..k+31.
// rowptr points at transposed weight row (column of B), contiguous in K.
__device__ __forceinline__ v16h load_b_frag(const _Float16* __restrict__ rowptr,
                                            int k, int lane) {
    int koff = k + ((lane >> 4) << 3);
    FragU f;
    f.h[0] = *(const v8h*)(rowptr + koff);
    f.h[1] = *(const v8h*)(rowptr + koff + 16);
    return f.v;
}

__device__ __forceinline__ float gelu_tanh(float v) {
    float u = 0.7978845608028654f * (v + 0.044715f * v * v * v);
    return 0.5f * v * (1.0f + tanhf(u));
}

// ---------------- small helper kernels ----------------
__global__ void k_init(int* __restrict__ counts) {
    if (threadIdx.x < NE) counts[threadIdx.x] = 0;
}

__global__ __launch_bounds__(256) void k_cvt(const float* __restrict__ src,
                                             _Float16* __restrict__ dst, size_t n) {
    size_t i = (size_t)blockIdx.x * blockDim.x + threadIdx.x;
    size_t stride = (size_t)gridDim.x * blockDim.x;
    for (; i < n; i += stride) dst[i] = (_Float16)src[i];
}

// batched transpose: src fp32 [R][C] -> dst f16 [C][R], blockIdx.z = batch
__global__ __launch_bounds__(256) void k_transpose(const float* __restrict__ src,
                                                   _Float16* __restrict__ dst,
                                                   int R, int C) {
    __shared__ float tile[32][33];
    const float* s = src + (size_t)blockIdx.z * R * C;
    _Float16*    d = dst + (size_t)blockIdx.z * R * C;
    int c0 = blockIdx.x * 32, r0 = blockIdx.y * 32;
    int tx = threadIdx.x & 31, ty = threadIdx.x >> 5;   // 32 x 8
#pragma unroll
    for (int j = 0; j < 32; j += 8)
        tile[ty + j][tx] = s[(size_t)(r0 + ty + j) * C + (c0 + tx)];
    __syncthreads();
#pragma unroll
    for (int j = 0; j < 32; j += 8)
        d[(size_t)(c0 + ty + j) * R + (r0 + tx)] = (_Float16)tile[tx][ty + j];
}

// ---------------- router: logits -> top2 -> softmax -> compacted lists ----------------
__global__ __launch_bounds__(256) void k_router(const float* __restrict__ x,
                                                const float* __restrict__ Wr,
                                                int* __restrict__ counts,
                                                int* __restrict__ tlist,
                                                int* __restrict__ rtE,
                                                int* __restrict__ rtP,
                                                float* __restrict__ rtG) {
    __shared__ float red[256 * NE];
    __shared__ float logits[NE];
    const int n = blockIdx.x, tid = threadIdx.x;
    const float* xr = x + (size_t)n * DM;
    float acc[NE];
#pragma unroll
    for (int e = 0; e < NE; ++e) acc[e] = 0.f;
    for (int d = tid; d < DM; d += 256) {
        float xv = xr[d];
#pragma unroll
        for (int e = 0; e < NE; ++e) acc[e] += xv * Wr[(size_t)e * DM + d];
    }
#pragma unroll
    for (int e = 0; e < NE; ++e) red[tid * NE + e] = acc[e];
    __syncthreads();
    if (tid < NE) {
        float s = 0.f;
        for (int t = 0; t < 256; ++t) s += red[t * NE + tid];
        logits[tid] = s;
    }
    __syncthreads();
    if (tid == 0) {
        // top-2 with jax.lax.top_k tie-breaking (lower index wins on ties)
        int e0 = 0; float v0 = logits[0];
        for (int e = 1; e < NE; ++e) if (logits[e] > v0) { v0 = logits[e]; e0 = e; }
        int e1 = (e0 == 0) ? 1 : 0; float v1 = logits[e1];
        for (int e = 0; e < NE; ++e) {
            if (e == e0 || e == e1) continue;
            if (logits[e] > v1) { v1 = logits[e]; e1 = e; }
        }
        float t  = expf(v1 - v0);           // softmax over the two top values
        float g0 = 1.f / (1.f + t);
        float g1 = t * g0;
        int p0 = atomicAdd(&counts[e0], 1); tlist[(size_t)e0 * NTOK + p0] = n;
        int p1 = atomicAdd(&counts[e1], 1); tlist[(size_t)e1 * NTOK + p1] = n;
        rtE[2 * n] = e0; rtE[2 * n + 1] = e1;
        rtP[2 * n] = p0; rtP[2 * n + 1] = p1;
        rtG[2 * n] = g0; rtG[2 * n + 1] = g1;
    }
}

__global__ void k_prefix(const int* __restrict__ counts, int* __restrict__ base) {
    if (threadIdx.x == 0) {
        int s = 0;
        for (int e = 0; e < NE; ++e) { base[e] = s; s += counts[e]; }
    }
}

// ---------------- pass 1: h = gelu(Xe @ W1[e] + b1[e]) ----------------
// block = 256 threads = 8 waves as 2(M) x 4(N); block tile 32 x 512; wave tile 16 x 128
__global__ __launch_bounds__(256) void k_mlp1(const _Float16* __restrict__ xh,   // [NTOK][DM]
                                              const _Float16* __restrict__ w1t,  // [NE][ID][DM]
                                              const float* __restrict__ b1,      // [NE][ID]
                                              const int* __restrict__ tlist,     // [NE][NTOK]
                                              const int* __restrict__ counts,
                                              const int* __restrict__ base,
                                              _Float16* __restrict__ hbuf) {     // [2*NTOK][ID]
    const int e   = blockIdx.z;
    const int cnt = counts[e];
    const int m0  = blockIdx.y * 32;
    if (m0 >= cnt) return;
    const int tid  = threadIdx.x;
    const int lane = tid & 31;
    const int wave = tid >> 5;
    const int mbase = m0 + (wave >> 2) * 16;
    const int nbase = blockIdx.x * (NTILES * 64) + (wave & 3) * (NTILES * 16);

    // A row pointer (gathered token row), clamped for ragged tail
    int r  = mbase + (lane & 15);
    int rc = r < cnt ? r : (cnt - 1);
    const _Float16* aptr = xh + (size_t)tlist[(size_t)e * NTOK + rc] * DM;

    const _Float16* bptr[NTILES];
#pragma unroll
    for (int nb = 0; nb < NTILES; ++nb)
        bptr[nb] = w1t + ((size_t)e * ID + (nbase + nb * 16 + (lane & 15))) * DM;

    v8f acc[NTILES] = {};
    for (int k = 0; k < DM; k += 32) {
        v16h a = load_a_frag(aptr, k, lane);
#pragma unroll
        for (int nb = 0; nb < NTILES; ++nb) {
            v16h b = load_b_frag(bptr[nb], k, lane);
            acc[nb] = __builtin_amdgcn_wmma_f32_16x16x32_f16(
                false, a, false, b, (short)0, acc[nb], false, false);
        }
    }

    const int hb = base[e];
#pragma unroll
    for (int nb = 0; nb < NTILES; ++nb) {
        int col = nbase + nb * 16 + (lane & 15);
        float bias = b1[(size_t)e * ID + col];
#pragma unroll
        for (int j = 0; j < 8; ++j) {
            int row = mbase + j + ((lane >> 4) << 3);
            float v = gelu_tanh(acc[nb][j] + bias);
            if (row < cnt)
                hbuf[(size_t)(hb + row) * ID + col] = (_Float16)v;
        }
    }
}

// ---------------- pass 2: eo = h @ W2[e] + b2[e] (fp32 out) ----------------
__global__ __launch_bounds__(256) void k_mlp2(const _Float16* __restrict__ hbuf,  // [2*NTOK][ID]
                                              const _Float16* __restrict__ w2t,   // [NE][DM][ID]
                                              const float* __restrict__ b2,       // [NE][DM]
                                              const int* __restrict__ counts,
                                              const int* __restrict__ base,
                                              float* __restrict__ eo) {           // [2*NTOK][DM]
    const int e   = blockIdx.z;
    const int cnt = counts[e];
    const int m0  = blockIdx.y * 32;
    if (m0 >= cnt) return;
    const int tid  = threadIdx.x;
    const int lane = tid & 31;
    const int wave = tid >> 5;
    const int mbase = m0 + (wave >> 2) * 16;
    const int nbase = blockIdx.x * (NTILES * 64) + (wave & 3) * (NTILES * 16);
    const int hb = base[e];

    int r  = mbase + (lane & 15);
    int rc = r < cnt ? r : (cnt - 1);
    const _Float16* aptr = hbuf + (size_t)(hb + rc) * ID;

    const _Float16* bptr[NTILES];
#pragma unroll
    for (int nb = 0; nb < NTILES; ++nb)
        bptr[nb] = w2t + ((size_t)e * DM + (nbase + nb * 16 + (lane & 15))) * ID;

    v8f acc[NTILES] = {};
    for (int k = 0; k < ID; k += 32) {
        v16h a = load_a_frag(aptr, k, lane);
#pragma unroll
        for (int nb = 0; nb < NTILES; ++nb) {
            v16h b = load_b_frag(bptr[nb], k, lane);
            acc[nb] = __builtin_amdgcn_wmma_f32_16x16x32_f16(
                false, a, false, b, (short)0, acc[nb], false, false);
        }
    }

#pragma unroll
    for (int nb = 0; nb < NTILES; ++nb) {
        int col = nbase + nb * 16 + (lane & 15);
        float bias = b2[(size_t)e * DM + col];
#pragma unroll
        for (int j = 0; j < 8; ++j) {
            int row = mbase + j + ((lane >> 4) << 3);
            if (row < cnt)
                eo[(size_t)(hb + row) * DM + col] = acc[nb][j] + bias;
        }
    }
}

// ---------------- final combine: out[n] = g0*eo[slot0] + g1*eo[slot1] ----------------
__global__ __launch_bounds__(256) void k_combine(const float* __restrict__ eo,
                                                 const int* __restrict__ rtE,
                                                 const int* __restrict__ rtP,
                                                 const float* __restrict__ rtG,
                                                 const int* __restrict__ base,
                                                 float* __restrict__ out) {
    const int n = blockIdx.x;
    const int e0 = rtE[2 * n], e1 = rtE[2 * n + 1];
    const size_t s0 = (size_t)(base[e0] + rtP[2 * n]) * DM;
    const size_t s1 = (size_t)(base[e1] + rtP[2 * n + 1]) * DM;
    const float g0 = rtG[2 * n], g1 = rtG[2 * n + 1];
    for (int d = threadIdx.x; d < DM; d += 256)
        out[(size_t)n * DM + d] = g0 * eo[s0 + d] + g1 * eo[s1 + d];
}

// ---------------- host-side launch ----------------
extern "C" void kernel_launch(void* const* d_in, const int* in_sizes, int n_in,
                              void* d_out, int out_size, void* d_ws, size_t ws_size,
                              hipStream_t stream) {
    const float* x  = (const float*)d_in[0];   // [2,2048,2048]
    const float* Wr = (const float*)d_in[1];   // [8,2048]
    const float* W1 = (const float*)d_in[2];   // [8,2048,1024]
    const float* b1 = (const float*)d_in[3];   // [8,1024]
    const float* W2 = (const float*)d_in[4];   // [8,1024,2048]
    const float* b2 = (const float*)d_in[5];   // [8,2048]
    float* out = (float*)d_out;

    char* ws = (char*)d_ws;
    size_t o_xh   = 0;                                   // f16  [NTOK][DM]      16 MB
    size_t o_w1t  = o_xh  + (size_t)NTOK * DM * 2;       // f16  [NE][ID][DM]    32 MB
    size_t o_w2t  = o_w1t + (size_t)NE * ID * DM * 2;    // f16  [NE][DM][ID]    32 MB
    size_t o_h    = o_w2t + (size_t)NE * DM * ID * 2;    // f16  [2*NTOK][ID]    16 MB
    size_t o_eo   = o_h   + (size_t)2 * NTOK * ID * 2;   // f32  [2*NTOK][DM]    64 MB
    size_t o_tl   = o_eo  + (size_t)2 * NTOK * DM * 4;   // int  [NE][NTOK]
    size_t o_cnt  = o_tl  + (size_t)NE * NTOK * 4;       // int  [NE]
    size_t o_base = o_cnt + 256;                         // int  [NE]
    size_t o_rtE  = o_base + 256;                        // int  [2*NTOK]
    size_t o_rtP  = o_rtE + (size_t)2 * NTOK * 4;        // int  [2*NTOK]
    size_t o_rtG  = o_rtP + (size_t)2 * NTOK * 4;        // f32  [2*NTOK]

    _Float16* xh   = (_Float16*)(ws + o_xh);
    _Float16* w1t  = (_Float16*)(ws + o_w1t);
    _Float16* w2t  = (_Float16*)(ws + o_w2t);
    _Float16* hbuf = (_Float16*)(ws + o_h);
    float*    eo   = (float*)   (ws + o_eo);
    int*      tl   = (int*)     (ws + o_tl);
    int*      cnt  = (int*)     (ws + o_cnt);
    int*      base = (int*)     (ws + o_base);
    int*      rtE  = (int*)     (ws + o_rtE);
    int*      rtP  = (int*)     (ws + o_rtP);
    float*    rtG  = (float*)   (ws + o_rtG);

    k_init<<<1, 32, 0, stream>>>(cnt);
    k_cvt<<<2048, 256, 0, stream>>>(x, xh, (size_t)NTOK * DM);
    k_transpose<<<dim3(ID / 32, DM / 32, NE), 256, 0, stream>>>(W1, w1t, DM, ID);
    k_transpose<<<dim3(DM / 32, ID / 32, NE), 256, 0, stream>>>(W2, w2t, ID, DM);
    k_router<<<NTOK, 256, 0, stream>>>(x, Wr, cnt, tl, rtE, rtP, rtG);
    k_prefix<<<1, 32, 0, stream>>>(cnt, base);
    // worst-case grids; blocks past counts[e] exit immediately
    k_mlp1<<<dim3(ID / (NTILES * 64), NTOK / 32, NE), 256, 0, stream>>>(xh, w1t, b1, tl, cnt, base, hbuf);
    k_mlp2<<<dim3(DM / (NTILES * 64), NTOK / 32, NE), 256, 0, stream>>>(hbuf, w2t, b2, cnt, base, eo);
    k_combine<<<NTOK, 256, 0, stream>>>(eo, rtE, rtP, rtG, base, out);
}
